// PeriodicSetTransformerEncoder_24704651887014
// MI455X (gfx1250) — matrix operand order repaired and checked by reading, not compiled, not run on hardware
//
#include <hip/hip_runtime.h>
#include <math.h>

typedef __bf16 bf16;
typedef __attribute__((ext_vector_type(16))) __bf16 v16bf;
typedef __attribute__((ext_vector_type(8)))  __bf16 v8bf;
typedef __attribute__((ext_vector_type(8)))  float   v8f;

static __device__ inline v8f zero8() {
  v8f z;
#pragma unroll
  for (int i = 0; i < 8; ++i) z[i] = 0.0f;
  return z;
}

// ---------------------------------------------------------------------------
// Optional CDNA5 async global->LDS copies (ASYNCcnt-tracked). Guarded by
// __has_builtin so either toolchain falls back to synchronous loads.
// The builtin expects non-const 'int __attribute__((vector_size(16)))*'
// pointers (global src, LDS dst); generic pointers with the exact pointee
// type are accepted and addrspace-cast by clang.
// ---------------------------------------------------------------------------
#if defined(__HIP_DEVICE_COMPILE__) && __has_builtin(__builtin_amdgcn_global_load_async_to_lds_b128)
#define ASYNC_LDS 1
typedef int b128v __attribute__((vector_size(16)));
static __device__ inline void async_copy16(const void* gsrc, void* ldst) {
  __builtin_amdgcn_global_load_async_to_lds_b128(
      (b128v*)(void*)(size_t)gsrc, (b128v*)ldst, 0, 0);
}
static __device__ inline void async_wait0() {
#if __has_builtin(__builtin_amdgcn_s_wait_asynccnt)
  __builtin_amdgcn_s_wait_asynccnt(0);
#else
  asm volatile("s_wait_asynccnt 0" ::: "memory");
#endif
}
#endif

// ---------------------------------------------------------------------------
// WMMA fragment loaders for V_WMMA_F32_16X16X32_BF16 (wave32).
// Per CDNA5 ISA 7.12.2:
//  A (16x32, 16-bit): lane = M + 16*khalf; lane holds K in [8*khalf,8*khalf+8)
//                     (VGPR0-3) and [16+8*khalf, 16+8*khalf+8) (VGPR4-7).
//  B (32x16, 16-bit): lane = N + 16*khalf; same K order as A (8*khalf.. and 16+8*khalf..).
//  C/D (16x16 f32):   lane = N + 16*(M/8); VGPR r holds M = r + 8*(lane/16).
// ---------------------------------------------------------------------------

// A-fragment: p -> element (row0, k0) of row-major matrix, leading dim ld.
static __device__ inline v16bf load_a_frag(const bf16* p, int ld) {
  int lane = threadIdx.x & 31;
  int ml = lane & 15;
  int kh = lane >> 4;
  const bf16* rp = p + (size_t)ml * ld + kh * 8;
  v8bf lo = *(const v8bf*)(rp);
  v8bf hi = *(const v8bf*)(rp + 16);
  v16bf a;
#pragma unroll
  for (int i = 0; i < 8; ++i) { a[i] = lo[i]; a[8 + i] = hi[i]; }
  return a;
}

// B-fragment where B[k][n] = S[n][k]; p -> element (n0, k0) of row-major S.
static __device__ inline v16bf load_bT_frag(const bf16* p, int ld) {
  int lane = threadIdx.x & 31;
  int nl = lane & 15;
  int kh = lane >> 4;
  const bf16* rp = p + (size_t)nl * ld + kh * 8;
  v8bf lo = *(const v8bf*)(rp);
  v8bf hi = *(const v8bf*)(rp + 16);
  v16bf b;
#pragma unroll
  for (int i = 0; i < 8; ++i) { b[i] = lo[i]; b[8 + i] = hi[i]; }
  return b;
}

static __device__ inline v8f wmma_bf16(v16bf a, v16bf b, v8f c) {
  return __builtin_amdgcn_wmma_f32_16x16x32_bf16(false, a, false, b, (short)0, c,
                                                 false, false);
}

// ---------------------------------------------------------------------------
// fp32 -> bf16 elementwise convert
// ---------------------------------------------------------------------------
__global__ void cvt_f32_bf16(const float* __restrict__ in, bf16* __restrict__ out, int n) {
  int i = blockIdx.x * 256 + threadIdx.x;
  if (i < n) out[i] = (bf16)in[i];
}

// ---------------------------------------------------------------------------
// 64-wide row reductions (one 64-thread block per row)
// ---------------------------------------------------------------------------
static __device__ inline float reduce64(float v, float* red, int c) {
  red[c] = v;
  __syncthreads();
#pragma unroll
  for (int s = 32; s > 0; s >>= 1) {
    if (c < s) red[c] += red[c + s];
    __syncthreads();
  }
  float r = red[0];
  __syncthreads();
  return r;
}

// LayerNorm over E=64; emits bf16 x_norm.
__global__ void ln_kernel(const float* __restrict__ x, const float* __restrict__ g,
                          const float* __restrict__ b, bf16* __restrict__ xn) {
  __shared__ float red[64];
  int row = blockIdx.x;
  int c = threadIdx.x;
  float v = x[(size_t)row * 64 + c];
  float mean = reduce64(v, red, c) * (1.0f / 64.0f);
  float d = v - mean;
  float var = reduce64(d * d, red, c) * (1.0f / 64.0f);
  float y = d * rsqrtf(var + 1e-5f) * g[c] + b[c];
  xn[(size_t)row * 64 + c] = (bf16)y;
}

// ---------------------------------------------------------------------------
// Generic bf16 WMMA GEMM: Y[M x N] = X[M x K] * W[N x K]^T + bias (+residual)
// block = 256 threads (8 waves, 4x2 wave grid of 16x16 tiles => 64x32 / block)
// ---------------------------------------------------------------------------
__global__ __launch_bounds__(256) void gemm_bf16(
    const bf16* __restrict__ X, const bf16* __restrict__ W,
    const float* __restrict__ bias, const float* __restrict__ residual,
    bf16* __restrict__ Yb, float* __restrict__ Yf, int M, int K, int N) {
  int wave = threadIdx.x >> 5;
  int lane = threadIdx.x & 31;
  int ml = lane & 15;
  int kh = lane >> 4;
  int m0 = blockIdx.x * 64 + (wave >> 1) * 16;
  int n0 = blockIdx.y * 32 + (wave & 1) * 16;

  v8f acc = zero8();
  const bf16* xp = X + (size_t)m0 * K;
  const bf16* wp = W + (size_t)n0 * K;
  for (int k0 = 0; k0 < K; k0 += 32) {
    v16bf a = load_a_frag(xp + k0, K);
    v16bf b = load_bT_frag(wp + k0, K);
    acc = wmma_bf16(a, b, acc);
  }

  int col = n0 + ml;
  float bv = bias[col];
#pragma unroll
  for (int r = 0; r < 8; ++r) {
    int row = m0 + r + 8 * kh;
    float v = acc[r] + bv;
    if (residual) v += residual[(size_t)row * N + col];
    if (Yf) Yf[(size_t)row * N + col] = v;
    if (Yb) Yb[(size_t)row * N + col] = (bf16)v;
  }
}

// ---------------------------------------------------------------------------
// Fused flash attention (heads looped, mean-over-heads + weight renorm folded):
//   num_i = sum_h sum_j p_hij * w_j * v_j ; den_i = sum_h sum_j p_hij * w_j
//   out_i = num_i / den_i
// grid.x = B * (N/64); block = 256 threads (8 waves); wave w owns D cols [32w,32w+32)
// ---------------------------------------------------------------------------
#define ITILE 64
#define JTILE 32

__global__ __launch_bounds__(256) void attention_kernel(
    const bf16* __restrict__ Q, const bf16* __restrict__ Kg,
    const bf16* __restrict__ V, const float* __restrict__ wts,
    bf16* __restrict__ O, int Nn) {
  const int D = 256, E = 64, H = 4;
  int nblk = Nn / ITILE;
  int b = blockIdx.x / nblk;
  int i0 = (blockIdx.x % nblk) * ITILE;
  int tid = threadIdx.x;
  int wave = tid >> 5;
  int lane = tid & 31;
  int ml = lane & 15;
  int kh = lane >> 4;

  __shared__ __align__(64) bf16 Qs[ITILE][E];       // 8 KB
  __shared__ __align__(64) bf16 Ks[JTILE][E];       // 4 KB
  __shared__ __align__(64) bf16 Vt[256][JTILE];     // 16 KB (V transposed: [d][j])
  __shared__ __align__(64) float Ss[ITILE][JTILE];  // 8 KB
  __shared__ __align__(64) bf16 Ps[ITILE][JTILE];   // 4 KB
  __shared__ float wj[JTILE], jb[JTILE];
  __shared__ float mrow[ITILE], lrow[ITILE], lwrow[ITILE], crow[ITILE], den[ITILE];

  v8f num[4][2];
#pragma unroll
  for (int ti = 0; ti < 4; ++ti)
#pragma unroll
    for (int td = 0; td < 2; ++td) num[ti][td] = zero8();
  if (tid < ITILE) den[tid] = 0.0f;

  for (int h = 0; h < H; ++h) {
    if (tid < ITILE) { mrow[tid] = -1e30f; lrow[tid] = 0.0f; lwrow[tid] = 0.0f; }
    v8f U[4][2];
#pragma unroll
    for (int ti = 0; ti < 4; ++ti)
#pragma unroll
      for (int td = 0; td < 2; ++td) U[ti][td] = zero8();

    // load Q tile [64 x 64] for this head
    {
      int r = tid >> 2;
      int c = (tid & 3) * 16;
      const bf16* src = Q + ((size_t)(b * Nn + i0 + r)) * D + h * E + c;
#ifdef ASYNC_LDS
      async_copy16(src, &Qs[r][c]);
      async_copy16(src + 8, &Qs[r][c + 8]);
      async_wait0();
#else
      *(v16bf*)&Qs[r][c] = *(const v16bf*)src;
#endif
    }
    __syncthreads();

    for (int j0 = 0; j0 < Nn; j0 += JTILE) {
      __syncthreads();  // previous iter's Ps/Vt consumers done before reload
      // K tile [32 x 64]  (async global->LDS when available)
      {
        int r = tid >> 3;
        int c = (tid & 7) * 8;
        const bf16* src = Kg + ((size_t)(b * Nn + j0 + r)) * D + h * E + c;
#ifdef ASYNC_LDS
        async_copy16(src, &Ks[r][c]);
#else
        *(v8bf*)&Ks[r][c] = *(const v8bf*)src;
#endif
        if (j0 + JTILE < Nn)
          __builtin_prefetch(Kg + ((size_t)(b * Nn + j0 + JTILE + r)) * D + h * E + c, 0, 3);
      }
      // V tile transposed into LDS: Vt[d][j]
      {
        int j = tid >> 3;
        int d0 = (tid & 7) * 32;
        const bf16* src = V + ((size_t)(b * Nn + j0 + j)) * D + d0;
#pragma unroll
        for (int cc = 0; cc < 32; cc += 8) {
          v8bf vv = *(const v8bf*)(src + cc);
#pragma unroll
          for (int c = 0; c < 8; ++c) Vt[d0 + cc + c][j] = vv[c];
        }
      }
      if (tid < JTILE) {
        float w = wts[b * Nn + j0 + tid];
        wj[tid] = w;
        jb[tid] = (w == 0.0f) ? -1e9f : 0.0f;  // key_padding_mask
      }
#ifdef ASYNC_LDS
      async_wait0();
#endif
      __syncthreads();

      // S = (Q K^T) * 1/sqrt(64) + mask ; 8 tiles, one per wave
      {
        int ti = wave >> 1, tj = wave & 1;
        v8f s = zero8();
        s = wmma_bf16(load_a_frag(&Qs[ti * 16][0], E), load_bT_frag(&Ks[tj * 16][0], E), s);
        s = wmma_bf16(load_a_frag(&Qs[ti * 16][32], E), load_bT_frag(&Ks[tj * 16][32], E), s);
        int col = tj * 16 + ml;
        float mb = jb[col];
#pragma unroll
        for (int r = 0; r < 8; ++r)
          Ss[ti * 16 + r + 8 * kh][col] = s[r] * 0.125f + mb;
      }
      __syncthreads();

      // online softmax row update; P~ = exp(s-m)*w stored bf16
      if (tid < ITILE) {
        int r = tid;
        float mx = -1e30f;
#pragma unroll
        for (int j = 0; j < JTILE; ++j) mx = fmaxf(mx, Ss[r][j]);
        float mnew = fmaxf(mrow[r], mx);
        float c = __expf(mrow[r] - mnew);
        float ps = 0.0f, pws = 0.0f;
#pragma unroll
        for (int j = 0; j < JTILE; ++j) {
          float e = __expf(Ss[r][j] - mnew);
          ps += e;
          float ew = e * wj[j];
          pws += ew;
          Ps[r][j] = (bf16)ew;
        }
        mrow[r] = mnew;
        lrow[r] = lrow[r] * c + ps;
        lwrow[r] = lwrow[r] * c + pws;
        crow[r] = c;
      }
      __syncthreads();

      // rescale U and accumulate U += P~ * V  (each wave: its 32 D-cols)
      {
        int dbase = wave * 32;
#pragma unroll
        for (int ti = 0; ti < 4; ++ti) {
          float cs[8];
#pragma unroll
          for (int r = 0; r < 8; ++r) cs[r] = crow[ti * 16 + r + 8 * kh];
#pragma unroll
          for (int td = 0; td < 2; ++td) {
#pragma unroll
            for (int r = 0; r < 8; ++r) U[ti][td][r] *= cs[r];
            U[ti][td] = wmma_bf16(load_a_frag(&Ps[ti * 16][0], JTILE),
                                  load_bT_frag(&Vt[dbase + td * 16][0], JTILE),
                                  U[ti][td]);
          }
        }
      }
    }  // j tiles

    // head epilogue: num += U / l ; den += lw / l
#pragma unroll
    for (int ti = 0; ti < 4; ++ti)
#pragma unroll
      for (int r = 0; r < 8; ++r) {
        float inv = 1.0f / lrow[ti * 16 + r + 8 * kh];
        num[ti][0][r] += U[ti][0][r] * inv;
        num[ti][1][r] += U[ti][1][r] * inv;
      }
    if (tid < ITILE) den[tid] += lwrow[tid] / lrow[tid];
    __syncthreads();
  }  // heads

  // out = num / den  (bf16, [B,N,D])
  {
    int dbase = wave * 32;
#pragma unroll
    for (int ti = 0; ti < 4; ++ti)
#pragma unroll
      for (int td = 0; td < 2; ++td)
#pragma unroll
        for (int r = 0; r < 8; ++r) {
          int row = ti * 16 + r + 8 * kh;
          int col = dbase + td * 16 + ml;
          O[((size_t)(b * Nn + i0 + row)) * D + col] = (bf16)(num[ti][td][r] / den[row]);
        }
  }
}

// ---------------------------------------------------------------------------
// Epilogue: out2 = softplus(LN(out1) @ ffn^T + b); out = LN(out1 + out2)
// one 64-thread block per row
// ---------------------------------------------------------------------------
__global__ void epilogue_kernel(const float* __restrict__ out1,
                                const float* __restrict__ g, const float* __restrict__ bt,
                                const float* __restrict__ ffn_w, const float* __restrict__ ffn_b,
                                float* __restrict__ out) {
  __shared__ float red[64];
  __shared__ float ln1[64];
  int row = blockIdx.x;
  int c = threadIdx.x;
  float x = out1[(size_t)row * 64 + c];
  float mean = reduce64(x, red, c) * (1.0f / 64.0f);
  float d = x - mean;
  float var = reduce64(d * d, red, c) * (1.0f / 64.0f);
  float l1 = d * rsqrtf(var + 1e-5f) * g[c] + bt[c];
  ln1[c] = l1;
  __syncthreads();
  float f = ffn_b[c];
  const float* wr = ffn_w + (size_t)c * 64;
#pragma unroll 8
  for (int k = 0; k < 64; ++k) f += ln1[k] * wr[k];
  float sp = (f > 20.0f) ? f : log1pf(__expf(f));
  float y = x + sp;
  float mean2 = reduce64(y, red, c) * (1.0f / 64.0f);
  float d2 = y - mean2;
  float var2 = reduce64(d2 * d2, red, c) * (1.0f / 64.0f);
  out[(size_t)row * 64 + c] = d2 * rsqrtf(var2 + 1e-5f) * g[c] + bt[c];
}

// ---------------------------------------------------------------------------
extern "C" void kernel_launch(void* const* d_in, const int* in_sizes, int n_in,
                              void* d_out, int out_size, void* d_ws, size_t ws_size,
                              hipStream_t stream) {
  (void)in_sizes; (void)n_in; (void)out_size; (void)ws_size;
  const float* x     = (const float*)d_in[0];
  const float* wts   = (const float*)d_in[1];
  const float* ln_g  = (const float*)d_in[2];
  const float* ln_b  = (const float*)d_in[3];
  const float* wq_w  = (const float*)d_in[4];
  const float* wq_b  = (const float*)d_in[5];
  const float* wk_w  = (const float*)d_in[6];
  const float* wk_b  = (const float*)d_in[7];
  const float* wv_w  = (const float*)d_in[8];
  const float* wv_b  = (const float*)d_in[9];
  const float* mq_w  = (const float*)d_in[10];
  const float* mq_b  = (const float*)d_in[11];
  const float* mk_w  = (const float*)d_in[12];
  const float* mk_b  = (const float*)d_in[13];
  const float* out_w = (const float*)d_in[14];
  const float* out_b = (const float*)d_in[15];
  const float* ffn_w = (const float*)d_in[16];
  const float* ffn_b = (const float*)d_in[17];
  float* out = (float*)d_out;

  const int B = 8, N = 2048, E = 64, D = 256;
  const int M = B * N;  // 16384

  char* ws = (char*)d_ws;
  size_t off = 0;
  auto alloc = [&](size_t bytes) -> void* {
    void* p = ws + off;
    off += (bytes + 255) & ~(size_t)255;
    return p;
  };
  bf16*  xn_b = (bf16*)alloc((size_t)M * E * 2);
  bf16*  tmp0 = (bf16*)alloc((size_t)M * D * 2);
  bf16*  qb   = (bf16*)alloc((size_t)M * D * 2);
  bf16*  kb   = (bf16*)alloc((size_t)M * D * 2);
  bf16*  vb   = (bf16*)alloc((size_t)M * D * 2);
  bf16*  attb = (bf16*)alloc((size_t)M * D * 2);
  float* out1 = (float*)alloc((size_t)M * E * 4);
  bf16*  wqb  = (bf16*)alloc((size_t)D * E * 2);
  bf16*  wkb  = (bf16*)alloc((size_t)D * E * 2);
  bf16*  wvb  = (bf16*)alloc((size_t)D * E * 2);
  bf16*  mqb  = (bf16*)alloc((size_t)D * D * 2);
  bf16*  mkb  = (bf16*)alloc((size_t)D * D * 2);
  bf16*  owb  = (bf16*)alloc((size_t)E * D * 2);

  // weight conversions fp32 -> bf16
  cvt_f32_bf16<<<(D * E + 255) / 256, 256, 0, stream>>>(wq_w, wqb, D * E);
  cvt_f32_bf16<<<(D * E + 255) / 256, 256, 0, stream>>>(wk_w, wkb, D * E);
  cvt_f32_bf16<<<(D * E + 255) / 256, 256, 0, stream>>>(wv_w, wvb, D * E);
  cvt_f32_bf16<<<(D * D + 255) / 256, 256, 0, stream>>>(mq_w, mqb, D * D);
  cvt_f32_bf16<<<(D * D + 255) / 256, 256, 0, stream>>>(mk_w, mkb, D * D);
  cvt_f32_bf16<<<(E * D + 255) / 256, 256, 0, stream>>>(out_w, owb, E * D);

  // x_norm (bf16)
  ln_kernel<<<M, 64, 0, stream>>>(x, ln_g, ln_b, xn_b);

  dim3 blk(256);
  // q = (xn @ wq^T + b) @ mq^T + b
  gemm_bf16<<<dim3(M / 64, D / 32), blk, 0, stream>>>(xn_b, wqb, wq_b, nullptr, tmp0, nullptr, M, E, D);
  gemm_bf16<<<dim3(M / 64, D / 32), blk, 0, stream>>>(tmp0, mqb, mq_b, nullptr, qb, nullptr, M, D, D);
  // k
  gemm_bf16<<<dim3(M / 64, D / 32), blk, 0, stream>>>(xn_b, wkb, wk_b, nullptr, tmp0, nullptr, M, E, D);
  gemm_bf16<<<dim3(M / 64, D / 32), blk, 0, stream>>>(tmp0, mkb, mk_b, nullptr, kb, nullptr, M, D, D);
  // v
  gemm_bf16<<<dim3(M / 64, D / 32), blk, 0, stream>>>(xn_b, wvb, wv_b, nullptr, vb, nullptr, M, E, D);

  // fused attention (mean-over-heads + weight renorm folded in)
  attention_kernel<<<B * (N / ITILE), 256, 0, stream>>>(qb, kb, vb, wts, attb, N);

  // out1 = x + att_out @ out_w^T + out_b  (fp32)
  gemm_bf16<<<dim3(M / 64, E / 32), blk, 0, stream>>>(attb, owb, out_b, x, nullptr, out1, M, D, E);

  // final LN/ffn/softplus/LN
  epilogue_kernel<<<M, 64, 0, stream>>>(out1, ln_g, ln_b, ffn_w, ffn_b, out);
}
